// MultiHeadAttention_63806034150218
// MI455X (gfx1250) — matrix operand order repaired and checked by reading, not compiled, not run on hardware
//
#include <hip/hip_runtime.h>
#include <hip/hip_bf16.h>
#include <math.h>

// MHA forward, MI455X (gfx1250, wave32).  v_wmma_f32_16x16x32_bf16 everywhere.
// - bf16 pre-pass: operands L2-resident (x 16MB, W^T 8MB << 192MB L2).
// - GEMMs: LDS-free, software-pipelined fragment loads under the WMMAs.
// - Flash attention: block-cooperative K/V tiles in LDS, double-buffered,
//   filled with GLOBAL_LOAD_ASYNC_TO_LDS_B128 (ASYNCcnt path).

typedef __bf16 bf16;
typedef __attribute__((ext_vector_type(16))) __bf16 v16bf;
typedef __attribute__((ext_vector_type(8)))  __bf16 v8bf;
typedef __attribute__((ext_vector_type(4)))  __bf16 v4bf;
typedef __attribute__((ext_vector_type(8)))  float  v8f;
typedef __attribute__((ext_vector_type(4)))  float  v4f;
typedef __attribute__((ext_vector_type(4)))  int    v4i;

#if __has_builtin(__builtin_amdgcn_global_load_async_to_lds_b128)
#define HAVE_ASYNC_LDS 1
typedef __attribute__((address_space(1))) v4i* as1_v4i_p;   // prints as "__device__"
typedef __attribute__((address_space(3))) v4i* as3_v4i_p;   // LDS
#endif

static __device__ __forceinline__ void copy16_g2l(const bf16* g, bf16* l) {
#ifdef HAVE_ASYNC_LDS
  __builtin_amdgcn_global_load_async_to_lds_b128((as1_v4i_p)g, (as3_v4i_p)l, 0, 0);
#else
  *(v8bf*)l = *(const v8bf*)g;            // global_load_b128 + ds_store_b128
#endif
}

static __device__ __forceinline__ void wait_stage() {
#ifdef HAVE_ASYNC_LDS
#if __has_builtin(__builtin_amdgcn_s_wait_asynccnt)
  __builtin_amdgcn_s_wait_asynccnt(0);
#else
  asm volatile("s_wait_asynccnt 0x0" ::: "memory");
#endif
#endif
}

static __device__ __forceinline__ v16bf cat16(v8bf lo, v8bf hi) {
  v16bf r;
#pragma unroll
  for (int i = 0; i < 8; ++i) { r[i] = lo[i]; r[i + 8] = hi[i]; }
  return r;
}

// A-fragment (16 x 32 bf16): lane = g*16+r holds row r,
// halves [g*8 .. g*8+7] and [16+g*8 .. 16+g*8+7]  (CDNA5 ISA 7.12.2)
static __device__ __forceinline__ v16bf load_afrag(const bf16* base, int stride, int kofs) {
  int lane = threadIdx.x & 31;
  int g = lane >> 4, r = lane & 15;
  const bf16* p = base + r * stride + kofs + g * 8;
  v8bf lo = *(const v8bf*)p;
  v8bf hi = *(const v8bf*)(p + 16);
  return cat16(lo, hi);
}

// B-fragment (32 x 16 bf16) from column-contiguous storage (B^T row-major):
// lane = g*16+n holds column n, K values [g*8 .. g*8+7] and [16+g*8 .. 16+g*8+7]
static __device__ __forceinline__ v16bf load_bfrag(const bf16* base, int stride, int kofs) {
  int lane = threadIdx.x & 31;
  int g = lane >> 4, n = lane & 15;
  const bf16* p = base + n * stride + kofs + g * 8;
  v8bf lo = *(const v8bf*)p;
  v8bf hi = *(const v8bf*)(p + 16);
  return cat16(lo, hi);
}

static __device__ __forceinline__ v8f wmma_bf16(v16bf a, v16bf b, v8f c) {
  return __builtin_amdgcn_wmma_f32_16x16x32_bf16(false, a, false, b, (short)0, c, false, false);
}

// ---------------------------------------------------------------------------
// Pre-pass A: fp32 -> bf16 streaming convert.
// ---------------------------------------------------------------------------
__global__ __launch_bounds__(256) void cvt_fp32_bf16(const float* __restrict__ src,
                                                     bf16* __restrict__ dst) {
  size_t i0 = ((size_t)blockIdx.x * 256 + threadIdx.x) * 8;
  v4f f0 = *(const v4f*)(src + i0);
  v4f f1 = *(const v4f*)(src + i0 + 4);
  v4bf b0, b1;
#pragma unroll
  for (int i = 0; i < 4; ++i) { b0[i] = (bf16)f0[i]; b1[i] = (bf16)f1[i]; }
  *(v4bf*)(dst + i0) = b0;
  *(v4bf*)(dst + i0 + 4) = b1;
}

// ---------------------------------------------------------------------------
// Pre-pass B: W[K=1024][N] fp32 -> Wt[N][1024] bf16, 64x64 LDS tile transpose.
// ---------------------------------------------------------------------------
__global__ __launch_bounds__(256) void transpose_to_bf16(const float* __restrict__ W,
                                                         bf16* __restrict__ Wt, int N) {
  __shared__ bf16 tile[64 * 72];
  int tid = threadIdx.x;
  int k0 = blockIdx.x * 64;
  int n0 = blockIdx.y * 64;
  float regs[16];
#pragma unroll
  for (int i = 0; i < 16; ++i) {
    int idx = tid + i * 256;
    int r = idx >> 6, c = idx & 63;
    regs[i] = W[(size_t)(k0 + r) * N + n0 + c];
  }
#pragma unroll
  for (int i = 0; i < 16; ++i) {
    int idx = tid + i * 256;
    int r = idx >> 6, c = idx & 63;
    tile[c * 72 + r] = (bf16)regs[i];
  }
  __syncthreads();
#pragma unroll
  for (int i = 0; i < 16; ++i) {
    int idx = tid + i * 256;
    int r = idx >> 6, c = idx & 63;
    Wt[(size_t)(n0 + r) * 1024 + k0 + c] = tile[r * 72 + c];
  }
}

// ---------------------------------------------------------------------------
// Software-pipelined LDS-free GEMM body: wave tile 32x64, 8 WMMA per K-step,
// next step's 12 b128 fragment loads issued under the current WMMAs.
// ---------------------------------------------------------------------------
#define GEMM_PIPELINED_LOOP(A0, A1, Bb, acc)                                  \
  v16bf a0 = load_afrag(A0, 1024, 0);                                         \
  v16bf a1 = load_afrag(A1, 1024, 0);                                         \
  v16bf b0 = load_bfrag(Bb, 1024, 0);                                         \
  v16bf b1 = load_bfrag(Bb + (size_t)16 * 1024, 1024, 0);                     \
  v16bf b2 = load_bfrag(Bb + (size_t)32 * 1024, 1024, 0);                     \
  v16bf b3 = load_bfrag(Bb + (size_t)48 * 1024, 1024, 0);                     \
  for (int k0 = 0; k0 < 1024; k0 += 32) {                                     \
    int kn = (k0 + 32) & 1023; /* wraps to 0 on last iter (discarded) */      \
    v16bf na0 = load_afrag(A0, 1024, kn);                                     \
    v16bf na1 = load_afrag(A1, 1024, kn);                                     \
    v16bf nb0 = load_bfrag(Bb, 1024, kn);                                     \
    v16bf nb1 = load_bfrag(Bb + (size_t)16 * 1024, 1024, kn);                 \
    v16bf nb2 = load_bfrag(Bb + (size_t)32 * 1024, 1024, kn);                 \
    v16bf nb3 = load_bfrag(Bb + (size_t)48 * 1024, 1024, kn);                 \
    acc[0][0] = wmma_bf16(a0, b0, acc[0][0]);                                 \
    acc[1][0] = wmma_bf16(a1, b0, acc[1][0]);                                 \
    acc[0][1] = wmma_bf16(a0, b1, acc[0][1]);                                 \
    acc[1][1] = wmma_bf16(a1, b1, acc[1][1]);                                 \
    acc[0][2] = wmma_bf16(a0, b2, acc[0][2]);                                 \
    acc[1][2] = wmma_bf16(a1, b2, acc[1][2]);                                 \
    acc[0][3] = wmma_bf16(a0, b3, acc[0][3]);                                 \
    acc[1][3] = wmma_bf16(a1, b3, acc[1][3]);                                 \
    a0 = na0; a1 = na1; b0 = nb0; b1 = nb1; b2 = nb2; b3 = nb3;               \
  }

// ---------------------------------------------------------------------------
// Kernel 1: QKV GEMM.  M=8192, K=1024, N=3072.  Block tile 128x128 (8 waves).
// Epilogue: +bias, Q*0.125 / K -> [B,H,T,64], V -> transposed [B,H,64,T].
// ---------------------------------------------------------------------------
__global__ __launch_bounds__(256) void mha_qkv_gemm(
    const bf16* __restrict__ xb, const bf16* __restrict__ Wt,
    const float* __restrict__ bqkv,
    bf16* __restrict__ q_ws, bf16* __restrict__ k_ws, bf16* __restrict__ v_ws) {
  int tid = threadIdx.x, lane = tid & 31, wid = tid >> 5;
  int wm = wid >> 1, wn = wid & 1;
  int m0 = blockIdx.x * 128 + wm * 32;
  int n0 = blockIdx.y * 128 + wn * 64;
  const bf16* A0 = xb + (size_t)m0 * 1024;
  const bf16* A1 = xb + (size_t)(m0 + 16) * 1024;
  const bf16* Bb = Wt + (size_t)n0 * 1024;

  v8f zero = {0.f, 0.f, 0.f, 0.f, 0.f, 0.f, 0.f, 0.f};
  v8f acc[2][4] = {{zero, zero, zero, zero}, {zero, zero, zero, zero}};
  GEMM_PIPELINED_LOOP(A0, A1, Bb, acc)

  int g = lane >> 4, col = lane & 15;
  int sec = (blockIdx.y * 128) >> 10;        // 0=q 1=k 2=v, uniform per block
#pragma unroll
  for (int am = 0; am < 2; ++am)
#pragma unroll
    for (int j = 0; j < 4; ++j) {
      int nn = n0 + j * 16 + col;
      float bias = bqkv[nn];
      int hh = (nn & 1023) >> 6;
      int d = nn & 63;
      int mrow0 = m0 + am * 16 + 8 * g;      // C layout: row = v + 8g
#pragma unroll
      for (int v = 0; v < 8; ++v) {
        int m = mrow0 + v;
        int bidx = m >> 11, t = m & 2047;
        float val = acc[am][j][v] + bias;
        if (sec == 0)
          q_ws[((size_t)(bidx * 16 + hh) * 2048 + t) * 64 + d] = (bf16)(val * 0.125f);
        else if (sec == 1)
          k_ws[((size_t)(bidx * 16 + hh) * 2048 + t) * 64 + d] = (bf16)val;
        else
          v_ws[((size_t)(bidx * 16 + hh) * 64 + d) * 2048 + t] = (bf16)val;
      }
    }
}

// ---------------------------------------------------------------------------
// Kernel 2: causal flash attention, flash-2 style.
// Block = 8 waves = 128 query rows of one head; K (32x64, pad72) and V^T
// (64x32, pad40) tiles staged cooperatively into double-buffered LDS via
// async-to-LDS copies; each wave computes its 16 rows with online softmax.
// ---------------------------------------------------------------------------
__global__ __launch_bounds__(256) void mha_flash_attn(
    const bf16* __restrict__ q_ws, const bf16* __restrict__ k_ws,
    const bf16* __restrict__ v_ws, bf16* __restrict__ o_ws) {
  __shared__ bf16 ldsK[2][32 * 72];
  __shared__ bf16 ldsVt[2][64 * 40];
  __shared__ bf16 ldsP[8 * 16 * 40];

  int tid = threadIdx.x, lane = tid & 31, wid = tid >> 5;
  int g = lane >> 4, n = lane & 15;
  int bh = blockIdx.x >> 4;
  int Q0 = (blockIdx.x & 15) * 128;
  int q0 = Q0 + wid * 16;

  const bf16* qbase = q_ws + ((size_t)bh * 2048 + q0) * 64;
  const bf16* kbase = k_ws + (size_t)bh * 2048 * 64;
  const bf16* vbase = v_ws + (size_t)bh * 64 * 2048;
  bf16* myP = ldsP + wid * 16 * 40;

  // cooperative stage of one 32-key block into LDS buffer `buf`
  int kr = tid >> 3, kc = (tid & 7) * 8;     // K tile: 32 rows x 64
  int vr = tid >> 2, vc = (tid & 3) * 8;     // V^T tile: 64 rows x 32
  auto stage = [&](int buf, int j0) {
    copy16_g2l(kbase + (size_t)(j0 + kr) * 64 + kc, &ldsK[buf][kr * 72 + kc]);
    copy16_g2l(vbase + (size_t)vr * 2048 + j0 + vc, &ldsVt[buf][vr * 40 + vc]);
  };

  v16bf qf0 = load_afrag(qbase, 64, 0);      // Q pre-scaled by 1/sqrt(Dh)
  v16bf qf1 = load_afrag(qbase, 64, 32);

  v8f zero = {0.f, 0.f, 0.f, 0.f, 0.f, 0.f, 0.f, 0.f};
  v8f acc[4] = {zero, zero, zero, zero};
  float mstate[8], lstate[8];
#pragma unroll
  for (int v = 0; v < 8; ++v) { mstate[v] = -INFINITY; lstate[v] = 0.f; }

  int nkb = Q0 / 32 + 4;                     // key blocks needed by this block
  stage(0, 0);
  for (int it = 0; it < nkb; ++it) {
    int j0 = it * 32;
    wait_stage();
    __syncthreads();                         // tile `it` fully in LDS
    if (it + 1 < nkb) stage((it + 1) & 1, j0 + 32);

    if (j0 <= q0 + 15) {                     // wave-uniform causal skip
      const bf16* kt = ldsK[it & 1];
      const bf16* vt = ldsVt[it & 1];
      v8f s0 = zero, s1 = zero;
      v16bf kbf;
      kbf = load_bfrag(kt, 72, 0);            s0 = wmma_bf16(qf0, kbf, s0);
      kbf = load_bfrag(kt, 72, 32);           s0 = wmma_bf16(qf1, kbf, s0);
      kbf = load_bfrag(kt + 16 * 72, 72, 0);  s1 = wmma_bf16(qf0, kbf, s1);
      kbf = load_bfrag(kt + 16 * 72, 72, 32); s1 = wmma_bf16(qf1, kbf, s1);

      float alpha[8];
#pragma unroll
      for (int v = 0; v < 8; ++v) {
        int qr = q0 + v + 8 * g;
        float a = s0[v];  if (j0 + n > qr)      a = -INFINITY;
        float b2 = s1[v]; if (j0 + 16 + n > qr) b2 = -INFINITY;
        float mx = fmaxf(a, b2);
#pragma unroll
        for (int off = 8; off >= 1; off >>= 1)
          mx = fmaxf(mx, __shfl_xor(mx, off, 32));
        float mnew = fmaxf(mstate[v], mx);
        float al = __expf(mstate[v] - mnew);
        float p0 = __expf(a - mnew);
        float p1 = __expf(b2 - mnew);
        float rs = p0 + p1;
#pragma unroll
        for (int off = 8; off >= 1; off >>= 1)
          rs += __shfl_xor(rs, off, 32);
        lstate[v] = lstate[v] * al + rs;
        mstate[v] = mnew;
        alpha[v] = al;
        int prow = (v + 8 * g) * 40;
        myP[prow + n]      = (bf16)p0;
        myP[prow + 16 + n] = (bf16)p1;
      }
#pragma unroll
      for (int dt = 0; dt < 4; ++dt)
#pragma unroll
        for (int v = 0; v < 8; ++v) acc[dt][v] *= alpha[v];

      asm volatile("s_wait_dscnt 0x0" ::: "memory");
      v16bf pf = load_afrag(myP, 40, 0);
#pragma unroll
      for (int dt = 0; dt < 4; ++dt) {
        v16bf vb = load_bfrag(vt + dt * 16 * 40, 40, 0);
        acc[dt] = wmma_bf16(pf, vb, acc[dt]);
      }
    }
  }

  int b = bh >> 4, h = bh & 15;
#pragma unroll
  for (int v = 0; v < 8; ++v) {
    float inv = 1.0f / lstate[v];
    size_t row = (size_t)(b * 2048 + q0 + v + 8 * g) * 1024 + h * 64;
#pragma unroll
    for (int dt = 0; dt < 4; ++dt)
      o_ws[row + dt * 16 + n] = (bf16)(acc[dt][v] * inv);
  }
}

// ---------------------------------------------------------------------------
// Kernel 3: output projection.  M=8192, K=1024, N=1024, fp32 out + bias.
// ---------------------------------------------------------------------------
__global__ __launch_bounds__(256) void mha_out_gemm(
    const bf16* __restrict__ o_ws, const bf16* __restrict__ Wt,
    const float* __restrict__ bout, float* __restrict__ out) {
  int tid = threadIdx.x, lane = tid & 31, wid = tid >> 5;
  int wm = wid >> 1, wn = wid & 1;
  int m0 = blockIdx.x * 128 + wm * 32;
  int n0 = blockIdx.y * 128 + wn * 64;
  const bf16* A0 = o_ws + (size_t)m0 * 1024;
  const bf16* A1 = o_ws + (size_t)(m0 + 16) * 1024;
  const bf16* Bb = Wt + (size_t)n0 * 1024;

  v8f zero = {0.f, 0.f, 0.f, 0.f, 0.f, 0.f, 0.f, 0.f};
  v8f acc[2][4] = {{zero, zero, zero, zero}, {zero, zero, zero, zero}};
  GEMM_PIPELINED_LOOP(A0, A1, Bb, acc)

  int g = lane >> 4, col = lane & 15;
#pragma unroll
  for (int am = 0; am < 2; ++am)
#pragma unroll
    for (int j = 0; j < 4; ++j) {
      int nn = n0 + j * 16 + col;
      float bias = bout[nn];
      int mrow0 = m0 + am * 16 + 8 * g;
#pragma unroll
      for (int v = 0; v < 8; ++v)
        out[(size_t)(mrow0 + v) * 1024 + nn] = acc[am][j][v] + bias;
    }
}

// ---------------------------------------------------------------------------
extern "C" void kernel_launch(void* const* d_in, const int* in_sizes, int n_in,
                              void* d_out, int out_size, void* d_ws, size_t ws_size,
                              hipStream_t stream) {
  (void)in_sizes; (void)n_in; (void)out_size; (void)ws_size;
  const float* x    = (const float*)d_in[0];
  const float* Wqkv = (const float*)d_in[1];
  const float* bqkv = (const float*)d_in[2];
  const float* Wout = (const float*)d_in[3];
  const float* bout = (const float*)d_in[4];
  float* out = (float*)d_out;

  bf16* xb   = (bf16*)d_ws;                       // 16 MB
  bf16* Wtq  = xb  + (size_t)8192 * 1024;         //  6 MB
  bf16* Wto  = Wtq + (size_t)3072 * 1024;         //  2 MB
  bf16* q_ws = Wto + (size_t)1024 * 1024;         // 16 MB
  bf16* k_ws = q_ws + (size_t)4 * 16 * 2048 * 64; // 16 MB
  bf16* v_ws = k_ws + (size_t)4 * 16 * 2048 * 64; // 16 MB
  bf16* o_ws = v_ws + (size_t)4 * 16 * 2048 * 64; // 16 MB

  cvt_fp32_bf16<<<dim3(4096), 256, 0, stream>>>(x, xb);
  transpose_to_bf16<<<dim3(16, 48), 256, 0, stream>>>(Wqkv, Wtq, 3072);
  transpose_to_bf16<<<dim3(16, 16), 256, 0, stream>>>(Wout, Wto, 1024);
  mha_qkv_gemm<<<dim3(64, 24), 256, 0, stream>>>(xb, Wtq, bqkv, q_ws, k_ws, v_ws);
  mha_flash_attn<<<dim3(1024), 256, 0, stream>>>(q_ws, k_ws, v_ws, o_ws);
  mha_out_gemm<<<dim3(64, 8), 256, 0, stream>>>(o_ws, Wto, bout, out);
}
